// SelfAttention_70815420776490
// MI455X (gfx1250) — compile-verified, matrix operand order checked by reading, not run on hardware
//
#include <hip/hip_runtime.h>

typedef __bf16 bf16_t;
typedef __attribute__((ext_vector_type(16))) __bf16 v16bf;
typedef __attribute__((ext_vector_type(8)))  __bf16 v8bf;
typedef __attribute__((ext_vector_type(8)))  float  v8f;

// Problem constants (match reference setup_inputs)
#define BATCH 8
#define CCH   512
#define NTOK  1024
#define NHEAD 8
#define HDIM  64

static __device__ __forceinline__ v16bf combine16(v8bf lo, v8bf hi) {
    v16bf r;
#pragma unroll
    for (int i = 0; i < 8; ++i) { r[i] = lo[i]; r[8 + i] = hi[i]; }
    return r;
}

// Load a WMMA 16-bit operand fragment (A-style layout) for one 16x32 tile.
// Per ISA: lanes 0-15 hold k = kofs+{0..7, 16..23}, lanes 16-31 hold
// k = kofs+{8..15, 24..31}; caller passes row = rowset + (lane&15).
static __device__ __forceinline__ v16bf ldfrag(const bf16_t* base, int stride,
                                               int row, int kofs, int lg) {
    const bf16_t* p = base + (size_t)row * stride + kofs + lg * 8;
    v8bf lo = *(const v8bf*)(p);
    v8bf hi = *(const v8bf*)(p + 16);
    return combine16(lo, hi);
}

static __device__ __forceinline__ v8f wmma_bf16(v16bf a, v16bf b, v8f c) {
    return __builtin_amdgcn_wmma_f32_16x16x32_bf16(false, a, false, b,
                                                   (short)0, c, false, false);
}

// CDNA5 async memory->LDS copy (ASYNCcnt-tracked), 16 bytes per lane.
// vdst = LDS byte address VGPR, vaddr = 64-bit global address VGPR pair.
static __device__ __forceinline__ void async_ld_b128(void* ldsp, const void* gp) {
    unsigned int lds = (unsigned int)(unsigned long long)(uintptr_t)ldsp;
    unsigned long long ga = (unsigned long long)(uintptr_t)gp;
    asm volatile("global_load_async_to_lds_b128 %0, %1, off"
                 :: "v"(lds), "v"(ga) : "memory");
}
static __device__ __forceinline__ void wait_async_le(int) {}
static __device__ __forceinline__ void wait_async3() {
    asm volatile("s_wait_asynccnt 0x3" ::: "memory");
}
static __device__ __forceinline__ void wait_async2() {
    asm volatile("s_wait_asynccnt 0x2" ::: "memory");
}
static __device__ __forceinline__ void wait_async0() {
    asm volatile("s_wait_asynccnt 0x0" ::: "memory");
}

// ---------------------------------------------------------------------------
// Kernel 0: tiled transpose with f32 -> bf16 convert: dst[j][i] = (bf16)src[i][j]
// src is R x Cc row-major f32; dst is Cc x R row-major bf16. blockDim (32,8).
// ---------------------------------------------------------------------------
__global__ __launch_bounds__(256)
void txp_cvt(const float* __restrict__ src, bf16_t* __restrict__ dst,
             int R, int Cc, long srcBatchStride, long dstBatchStride) {
    __shared__ float tile[32][33];
    src += (long)blockIdx.z * srcBatchStride;
    dst += (long)blockIdx.z * dstBatchStride;
    int j0 = blockIdx.x * 32;   // column tile in src
    int i0 = blockIdx.y * 32;   // row tile in src
    int tx = threadIdx.x, ty = threadIdx.y;
#pragma unroll
    for (int ii = 0; ii < 4; ++ii)
        tile[ty + 8 * ii][tx] = src[(size_t)(i0 + ty + 8 * ii) * Cc + j0 + tx];
    __syncthreads();
#pragma unroll
    for (int ii = 0; ii < 4; ++ii)
        dst[(size_t)(j0 + ty + 8 * ii) * R + i0 + tx] = (bf16_t)tile[tx][ty + 8 * ii];
}

// ---------------------------------------------------------------------------
// Tiled GEMM with double-buffered async global->LDS staging:
// D[1024x512] = A[1024x512](bf16,row-major) * Bt^T, Bt[512x512] row-major.
// Block = 256 threads (8 waves), tile 128(M) x 64(N), K-steps of 32.
// mode: 0/1 -> bf16 [bh][N][d] (Q or K); 2 -> bf16 [bh][d][N] (V^T);
//       3 -> f32 [b][N][C] (pre-residual out)
// ---------------------------------------------------------------------------
__global__ __launch_bounds__(256)
void gemm_bf16(const bf16_t* __restrict__ A, const bf16_t* __restrict__ Bt,
               void* __restrict__ OutP, int mode) {
    __shared__ __align__(16) bf16_t As[2][128][40];  // 2 x 10240 B
    __shared__ __align__(16) bf16_t Bs[2][64][40];   // 2 x  5120 B

    const int b   = blockIdx.z;
    const int m0  = blockIdx.x * 128;
    const int n0  = blockIdx.y * 64;
    const bf16_t* Ab = A + (size_t)b * NTOK * CCH;

    const int tid  = threadIdx.x;
    const int wid  = tid >> 5;
    const int lane = tid & 31;
    const int lr   = lane & 15;
    const int lg   = lane >> 4;
    const int srow = tid >> 2;          // 0..63
    const int scol = (tid & 3) * 8;     // 0,8,16,24

    v8f acc[4] = {};

    auto stage = [&](int buf, int kt) {
        const int kofs = kt * 32;
        async_ld_b128(&As[buf][srow][scol],
                      &Ab[(size_t)(m0 + srow) * CCH + kofs + scol]);
        async_ld_b128(&As[buf][srow + 64][scol],
                      &Ab[(size_t)(m0 + srow + 64) * CCH + kofs + scol]);
        async_ld_b128(&Bs[buf][srow][scol],
                      &Bt[(size_t)(n0 + srow) * CCH + kofs + scol]);
    };
    auto compute = [&](int buf) {
        v16bf af = ldfrag(&As[buf][0][0], 40, wid * 16 + lr, 0, lg);
#pragma unroll
        for (int nb = 0; nb < 4; ++nb) {
            v16bf bfr = ldfrag(&Bs[buf][0][0], 40, nb * 16 + lr, 0, lg);
            acc[nb] = wmma_bf16(af, bfr, acc[nb]);
        }
    };

    stage(0, 0);
    for (int kt = 0; kt < CCH / 32 - 1; ++kt) {
        stage((kt + 1) & 1, kt + 1);     // prefetch next tile (other buffer)
        wait_async3();                   // 3 current done (3 next in flight)
        __syncthreads();
        compute(kt & 1);
        __syncthreads();                 // protect buffer before next overwrite
    }
    wait_async0();
    __syncthreads();
    compute((CCH / 32 - 1) & 1);

    // D layout: component r -> M = r + 8*lg, N = lr (within 16x16 tile)
    const int head = n0 >> 6;  // 64 output cols per head
    if (mode <= 1) {
        bf16_t* O = (bf16_t*)OutP;  // [B*NHEAD][NTOK][HDIM]
        const size_t base = (size_t)(b * NHEAD + head) * NTOK;
#pragma unroll
        for (int nb = 0; nb < 4; ++nb) {
            const int dd = (nb * 16 + lr) & (HDIM - 1);
#pragma unroll
            for (int r = 0; r < 8; ++r) {
                const int row = m0 + wid * 16 + r + 8 * lg;
                O[(base + row) * HDIM + dd] = (bf16_t)acc[nb][r];
            }
        }
    } else if (mode == 2) {
        bf16_t* O = (bf16_t*)OutP;  // [B*NHEAD][HDIM][NTOK]  (V transposed)
        const size_t base = (size_t)(b * NHEAD + head) * HDIM;
#pragma unroll
        for (int nb = 0; nb < 4; ++nb) {
            const int dd = (nb * 16 + lr) & (HDIM - 1);
#pragma unroll
            for (int r = 0; r < 8; ++r) {
                const int row = m0 + wid * 16 + r + 8 * lg;
                O[(base + dd) * NTOK + row] = (bf16_t)acc[nb][r];
            }
        }
    } else {
        float* O = (float*)OutP;  // [B][NTOK][CCH] f32
#pragma unroll
        for (int nb = 0; nb < 4; ++nb) {
#pragma unroll
            for (int r = 0; r < 8; ++r) {
                const int row = m0 + wid * 16 + r + 8 * lg;
                O[((size_t)b * NTOK + row) * CCH + n0 + nb * 16 + lr] = acc[nb][r];
            }
        }
    }
}

// ---------------------------------------------------------------------------
// Flash attention: grid (NTOK/128, B*NHEAD), block 256 (8 waves).
// Each wave owns 16 query rows; the block cooperatively streams 32 keys/iter
// through double-buffered LDS (async copies, shared by all 8 waves).
// Q,K: [bh][NTOK][HDIM] bf16; Vt: [bh][HDIM][NTOK] bf16.
// Output attn (merged heads) bf16 [B][NTOK][CCH].
// ---------------------------------------------------------------------------
__global__ __launch_bounds__(256)
void attn_flash(const bf16_t* __restrict__ Q, const bf16_t* __restrict__ K,
                const bf16_t* __restrict__ Vt, bf16_t* __restrict__ attnOut) {
    __shared__ __align__(16) bf16_t Ks[2][32][80];  // 2 x 5120 B (keys x d)
    __shared__ __align__(16) bf16_t Vs[2][64][40];  // 2 x 5120 B (d x keys)
    __shared__ __align__(16) bf16_t Pl[8][16][32];  // per-wave P staging 8 KB

    const int bh   = blockIdx.y;
    const int b    = bh >> 3;
    const int head = bh & 7;
    const int tid  = threadIdx.x;
    const int wid  = tid >> 5;
    const int lane = tid & 31;
    const int lr   = lane & 15;
    const int lg   = lane >> 4;
    const int q0   = blockIdx.x * 128 + wid * 16;  // first query row of wave

    const bf16_t* Qb = Q  + (size_t)bh * NTOK * HDIM;
    const bf16_t* Kb = K  + (size_t)bh * NTOK * HDIM;
    const bf16_t* Vb = Vt + (size_t)bh * HDIM * NTOK;

    // Q fragments held in registers (16 rows x 64 k)
    const v16bf qf0 = ldfrag(Qb, HDIM, q0 + lr, 0, lg);
    const v16bf qf1 = ldfrag(Qb, HDIM, q0 + lr, 32, lg);

    v8f Oa[4] = {};
    float mrun[8], lrun[8];
#pragma unroll
    for (int r = 0; r < 8; ++r) { mrun[r] = -1e30f; lrun[r] = 0.0f; }

    const float SC = 0.125f * 1.44269504088896f;  // 1/sqrt(64) * log2(e)

    auto stageKV = [&](int buf, int j) {
        const int jk = j * 32;
        // K tile: 32 rows x 64 cols -> one b128 per thread
        async_ld_b128(&Ks[buf][tid >> 3][(tid & 7) * 8],
                      &Kb[(size_t)(jk + (tid >> 3)) * HDIM + (tid & 7) * 8]);
        // V^T tile: 64 rows x 32 cols -> one b128 per thread
        async_ld_b128(&Vs[buf][tid >> 2][(tid & 3) * 8],
                      &Vb[(size_t)(tid >> 2) * NTOK + jk + (tid & 3) * 8]);
    };

    auto compute = [&](int buf) {
        // scores: S0 = Q * K[0..15]^T, S1 = Q * K[16..31]^T (keys from LDS)
        v16bf k00 = ldfrag(&Ks[buf][0][0], 80, lr,      0,  lg);
        v16bf k01 = ldfrag(&Ks[buf][0][0], 80, lr,      32, lg);
        v16bf k10 = ldfrag(&Ks[buf][0][0], 80, 16 + lr, 0,  lg);
        v16bf k11 = ldfrag(&Ks[buf][0][0], 80, 16 + lr, 32, lg);
        v8f s0 = {}; s0 = wmma_bf16(qf0, k00, s0); s0 = wmma_bf16(qf1, k01, s0);
        v8f s1 = {}; s1 = wmma_bf16(qf0, k10, s1); s1 = wmma_bf16(qf1, k11, s1);

        float corr[8];
#pragma unroll
        for (int r = 0; r < 8; ++r) {
            float a  = s0[r] * SC;
            float bb = s1[r] * SC;
            float t  = fmaxf(a, bb);
            t = fmaxf(t, __shfl_xor(t, 1, 32));
            t = fmaxf(t, __shfl_xor(t, 2, 32));
            t = fmaxf(t, __shfl_xor(t, 4, 32));
            t = fmaxf(t, __shfl_xor(t, 8, 32));   // rowmax over 16-lane group
            const float mn = fmaxf(mrun[r], t);
            const float c  = exp2f(mrun[r] - mn);
            const float pa = exp2f(a - mn);
            const float pb = exp2f(bb - mn);
            float rs = pa + pb;
            rs += __shfl_xor(rs, 1, 32);
            rs += __shfl_xor(rs, 2, 32);
            rs += __shfl_xor(rs, 4, 32);
            rs += __shfl_xor(rs, 8, 32);
            lrun[r] = lrun[r] * c + rs;
            mrun[r] = mn;
            corr[r] = c;
            // D-layout -> row-major P in per-wave LDS (16 rows x 32 keys)
            Pl[wid][r + 8 * lg][lr]      = (bf16_t)pa;
            Pl[wid][r + 8 * lg][16 + lr] = (bf16_t)pb;
        }
#pragma unroll
        for (int nb = 0; nb < 4; ++nb)
#pragma unroll
            for (int r = 0; r < 8; ++r) Oa[nb][r] *= corr[r];

        asm volatile("s_wait_dscnt 0" ::: "memory");  // P writes -> reads
        v16bf pf = ldfrag(&Pl[wid][0][0], 32, lr, 0, lg);
#pragma unroll
        for (int nb = 0; nb < 4; ++nb) {
            v16bf vf = ldfrag(&Vs[buf][0][0], 40, nb * 16 + lr, 0, lg);
            Oa[nb] = wmma_bf16(pf, vf, Oa[nb]);
        }
    };

    stageKV(0, 0);
    for (int j = 0; j < NTOK / 32 - 1; ++j) {
        stageKV((j + 1) & 1, j + 1);    // prefetch next K/V tile
        wait_async2();                  // current tile's 2 copies done
        __syncthreads();
        compute(j & 1);
        __syncthreads();                // buffer safe before next overwrite
    }
    wait_async0();
    __syncthreads();
    compute((NTOK / 32 - 1) & 1);

    // normalize and store merged-head attn bf16 [B][NTOK][CCH]
    float inv[8];
#pragma unroll
    for (int r = 0; r < 8; ++r) inv[r] = 1.0f / lrun[r];
#pragma unroll
    for (int nb = 0; nb < 4; ++nb) {
        const int col = head * HDIM + nb * 16 + lr;
#pragma unroll
        for (int r = 0; r < 8; ++r) {
            const int row = q0 + r + 8 * lg;
            attnOut[((size_t)b * NTOK + row) * CCH + col] = (bf16_t)(Oa[nb][r] * inv[r]);
        }
    }
}

// ---------------------------------------------------------------------------
// Epilogue: y[b][c][n] = gamma * out[b][n][c] + x[b][c][n]  (LDS transpose)
// blockDim (32,8); grid (NTOK/32, CCH/32, B)
// ---------------------------------------------------------------------------
__global__ __launch_bounds__(256)
void epilogue(const float* __restrict__ outf, const float* __restrict__ x,
              const float* __restrict__ gamma, float* __restrict__ y) {
    __shared__ float tile[32][33];
    const int b  = blockIdx.z;
    const int n0 = blockIdx.x * 32;
    const int c0 = blockIdx.y * 32;
    const int tx = threadIdx.x, ty = threadIdx.y;
#pragma unroll
    for (int ii = 0; ii < 4; ++ii)
        tile[ty + 8 * ii][tx] = outf[((size_t)b * NTOK + n0 + ty + 8 * ii) * CCH + c0 + tx];
    __syncthreads();
    const float g = gamma[0];
#pragma unroll
    for (int ii = 0; ii < 4; ++ii) {
        const int c = c0 + ty + 8 * ii;
        const size_t idx = ((size_t)b * CCH + c) * NTOK + n0 + tx;
        y[idx] = g * tile[tx][ty + 8 * ii] + x[idx];
    }
}

// ---------------------------------------------------------------------------
extern "C" void kernel_launch(void* const* d_in, const int* in_sizes, int n_in,
                              void* d_out, int out_size, void* d_ws, size_t ws_size,
                              hipStream_t stream) {
    (void)in_sizes; (void)n_in; (void)out_size; (void)ws_size;
    const float* x     = (const float*)d_in[0];
    const float* Wq    = (const float*)d_in[1];
    const float* Wk    = (const float*)d_in[2];
    const float* Wv    = (const float*)d_in[3];
    const float* Wo    = (const float*)d_in[4];
    const float* gamma = (const float*)d_in[5];
    float* y = (float*)d_out;

    char* ws = (char*)d_ws;
    const size_t MB = 1u << 20;
    const size_t WELEM = (size_t)CCH * CCH;                // 262144
    bf16_t* Wt   = (bf16_t*)(ws);                          // 4 x 0.5 MB
    bf16_t* xb   = (bf16_t*)(ws + 2 * MB);                 // 8 MB  [B][N][C]
    bf16_t* Qb   = (bf16_t*)(ws + 10 * MB);                // 8 MB  [bh][N][d]
    bf16_t* Kb   = (bf16_t*)(ws + 18 * MB);                // 8 MB  [bh][N][d]
    bf16_t* Vt   = (bf16_t*)(ws + 26 * MB);                // 8 MB  [bh][d][N]
    bf16_t* attn = xb;                                     // reuse (xb dead after QKV)
    float*  outf = (float*)(ws + 10 * MB);                 // 16 MB, reuse Q+K region

    dim3 tb(32, 8, 1);
    // weights -> transposed bf16
    txp_cvt<<<dim3(16, 16, 1), tb, 0, stream>>>(Wq, Wt + 0 * WELEM, CCH, CCH, 0, 0);
    txp_cvt<<<dim3(16, 16, 1), tb, 0, stream>>>(Wk, Wt + 1 * WELEM, CCH, CCH, 0, 0);
    txp_cvt<<<dim3(16, 16, 1), tb, 0, stream>>>(Wv, Wt + 2 * WELEM, CCH, CCH, 0, 0);
    txp_cvt<<<dim3(16, 16, 1), tb, 0, stream>>>(Wo, Wt + 3 * WELEM, CCH, CCH, 0, 0);
    // x [B][C][N] -> xb [B][N][C] bf16
    txp_cvt<<<dim3(NTOK / 32, CCH / 32, BATCH), tb, 0, stream>>>(
        x, xb, CCH, NTOK, (long)CCH * NTOK, (long)NTOK * CCH);

    dim3 gg(NTOK / 128, CCH / 64, BATCH);
    gemm_bf16<<<gg, 256, 0, stream>>>(xb, Wt + 0 * WELEM, Qb, 0);
    gemm_bf16<<<gg, 256, 0, stream>>>(xb, Wt + 1 * WELEM, Kb, 1);
    gemm_bf16<<<gg, 256, 0, stream>>>(xb, Wt + 2 * WELEM, Vt, 2);

    attn_flash<<<dim3(NTOK / 128, BATCH * NHEAD), 256, 0, stream>>>(Qb, Kb, Vt, attn);

    gemm_bf16<<<gg, 256, 0, stream>>>(attn, Wt + 3 * WELEM, outf, 3);

    epilogue<<<dim3(NTOK / 32, CCH / 32, BATCH), tb, 0, stream>>>(outf, x, gamma, y);
}
